// mv_dense_61323543052434
// MI455X (gfx1250) — matrix operand order repaired and checked by reading, not compiled, not run on hardware
//
#include <hip/hip_runtime.h>
#include <stdint.h>

#define NUM_VARI 32
#define TWO_D    256
#define DIM_TO   256
#define BATCH    4096

typedef __bf16 bf16_t;
typedef __attribute__((ext_vector_type(16))) __bf16 v16bf;
typedef __attribute__((ext_vector_type(8)))  __bf16 v8bf;
typedef __attribute__((ext_vector_type(4)))  __bf16 v4bf;
typedef __attribute__((ext_vector_type(8)))  float  v8f;

constexpr int BK   = 32;
constexpr int LDSB = BK + 8;   // 40 elems -> 80B row stride: conflict-free b128 LDS reads

__device__ __forceinline__ bf16_t f2bf(float f) {
  unsigned u = __float_as_uint(f);
  unsigned r = (u + 0x7FFFu + ((u >> 16) & 1u)) >> 16;   // round-to-nearest-even
  return __builtin_bit_cast(bf16_t, (unsigned short)r);
}
__device__ __forceinline__ float bf2f(bf16_t h) {
  unsigned u = ((unsigned)__builtin_bit_cast(unsigned short, h)) << 16;
  return __uint_as_float(u);
}
// softplus with hardware transcendentals: v_exp_f32 / v_log_f32 (base-2)
__device__ __forceinline__ float softplus_f(float x) {
  float ax = fabsf(x);
  float t  = __builtin_amdgcn_exp2f(ax * -1.44269504088896340736f);   // e^-|x|
  return fmaxf(x, 0.0f) + 0.69314718055994530942f * __builtin_amdgcn_logf(1.0f + t);
}
__device__ __forceinline__ v16bf cat8(v8bf a, v8bf b) {
  return __builtin_shufflevector(a, b, 0,1,2,3,4,5,6,7,8,9,10,11,12,13,14,15);
}
// CDNA5 async global->LDS copy (ASYNCcnt path). LDS offset = low 32 bits of flat addr.
__device__ __forceinline__ void async_ld_b128(void* lds_dst, const void* gsrc) {
  unsigned l = (unsigned)(uintptr_t)lds_dst;
  unsigned long long g = (unsigned long long)(uintptr_t)gsrc;
  asm volatile("global_load_async_to_lds_b128 %0, %1, off" :: "v"(l), "v"(g) : "memory");
}
__device__ __forceinline__ void wait_async0() {
  asm volatile("s_wait_asynccnt 0x0" ::: "memory");
}

// ---- prologue: split W (f32 [v][k][o]) into bf16 hi/lo planes, transposed [v][o][k] ----
__global__ __launch_bounds__(256)
void convert_w_kernel(const float* __restrict__ w,
                      bf16_t* __restrict__ whi, bf16_t* __restrict__ wlo) {
  const int idx = blockIdx.x * blockDim.x + threadIdx.x;  // 0..8191 = v*256 + o
  const int v = idx >> 8, o = idx & 255;
  const float* src = w + (size_t)v * TWO_D * DIM_TO + o;          // stride DIM_TO along k
  bf16_t* dh = whi + ((size_t)v * DIM_TO + o) * TWO_D;
  bf16_t* dl = wlo + ((size_t)v * DIM_TO + o) * TWO_D;
  for (int k4 = 0; k4 < TWO_D; k4 += 4) {
    v4bf h4, l4;
    #pragma unroll
    for (int j = 0; j < 4; ++j) {
      const float f = src[(size_t)(k4 + j) * DIM_TO];
      const bf16_t hh = f2bf(f);
      h4[j] = hh;
      l4[j] = f2bf(f - bf2f(hh));
    }
    *(v4bf*)&dh[k4] = h4;
    *(v4bf*)&dl[k4] = l4;
  }
}

// ---- main GEMM: 128x256 block, 512 threads (16 waves, 4Mx4N), bf16x3 WMMA ----
__global__ __launch_bounds__(512)
void gemm_softplus_fast(const float* __restrict__ x,
                        const bf16_t* __restrict__ whi,
                        const bf16_t* __restrict__ wlo,
                        const float* __restrict__ bias,
                        float* __restrict__ out) {
  constexpr int BM = 128, BN = 256;
  __shared__ __align__(16) bf16_t sAhi[BM * LDSB];
  __shared__ __align__(16) bf16_t sAlo[BM * LDSB];
  __shared__ __align__(16) bf16_t sBhi[BN * LDSB];   // [o][k]
  __shared__ __align__(16) bf16_t sBlo[BN * LDSB];

  const int t  = threadIdx.x;
  const int v  = blockIdx.z;
  const int m0 = blockIdx.x * BM;

  const float*  xg   = x   + ((size_t)v * BATCH + m0) * TWO_D;
  const bf16_t* bh_g = whi + (size_t)v * DIM_TO * TWO_D;
  const bf16_t* bl_g = wlo + (size_t)v * DIM_TO * TWO_D;

  const int lane = t & 31;
  const int h    = lane >> 4;       // lane half -> K-half per ISA layout
  const int ln   = lane & 15;
  const int wv   = t >> 5;          // 0..15
  const int wm   = (wv >> 2) * 32;  // 0,32,64,96
  const int wn   = (wv & 3) * 64;   // 0,64,128,192

  const int ar = t >> 3;            // 0..63
  const int ac = (t & 7) * 4;

  v8f acc[2][4] = {};

  for (int k0 = 0; k0 < TWO_D; k0 += BK) {
    __syncthreads();

    // ---- B tile: async-copy pre-converted bf16 rows into LDS (2048 x 16B tasks) ----
    #pragma unroll
    for (int i = 0; i < 4; ++i) {
      const int task  = t + i * 512;        // 0..2047
      const int plane = task >> 10;         // 0: hi, 1: lo
      const int row   = (task >> 2) & 255;  // o
      const int kq    = task & 3;           // 16B quarter of the 64B row chunk
      const bf16_t* g = (plane ? bl_g : bh_g) + (size_t)row * TWO_D + (k0 + kq * 8);
      bf16_t*       l = (plane ? sBlo : sBhi) + row * LDSB + kq * 8;
      async_ld_b128(l, g);
    }

    // ---- A tile: load f32, split to bf16 hi/lo, store to LDS ----
    #pragma unroll
    for (int p = 0; p < 2; ++p) {
      const int row = p * 64 + ar;
      const float4 f = *(const float4*)(xg + (size_t)row * TWO_D + (k0 + ac));
      const float fx[4] = {f.x, f.y, f.z, f.w};
      v4bf hi4, lo4;
      #pragma unroll
      for (int j = 0; j < 4; ++j) {
        const bf16_t hh = f2bf(fx[j]);
        hi4[j] = hh;
        lo4[j] = f2bf(fx[j] - bf2f(hh));
      }
      *(v4bf*)&sAhi[row * LDSB + ac] = hi4;
      *(v4bf*)&sAlo[row * LDSB + ac] = lo4;
    }

    wait_async0();
    __syncthreads();

    // ---- fragment loads (ISA 16-bit A / B lane layouts) ----
    v16bf ah[2], al[2], bh[4], bl[4];
    #pragma unroll
    for (int mi = 0; mi < 2; ++mi) {
      const int row = wm + mi * 16 + ln;
      const bf16_t* ph = &sAhi[row * LDSB];
      const bf16_t* pl = &sAlo[row * LDSB];
      ah[mi] = cat8(*(const v8bf*)(ph + h * 8), *(const v8bf*)(ph + 16 + h * 8));
      al[mi] = cat8(*(const v8bf*)(pl + h * 8), *(const v8bf*)(pl + 16 + h * 8));
    }
    #pragma unroll
    for (int ni = 0; ni < 4; ++ni) {
      const int col = wn + ni * 16 + ln;
      const bf16_t* ph = &sBhi[col * LDSB];
      const bf16_t* pl = &sBlo[col * LDSB];
      bh[ni] = cat8(*(const v8bf*)(ph + h * 8), *(const v8bf*)(ph + 16 + h * 8));
      bl[ni] = cat8(*(const v8bf*)(pl + h * 8), *(const v8bf*)(pl + 16 + h * 8));
    }

    // ---- bf16x3 split precision: hi*hi + hi*lo + lo*hi ----
    #pragma unroll
    for (int mi = 0; mi < 2; ++mi) {
      #pragma unroll
      for (int ni = 0; ni < 4; ++ni) {
        acc[mi][ni] = __builtin_amdgcn_wmma_f32_16x16x32_bf16(
            false, ah[mi], false, bh[ni], (short)0, acc[mi][ni], false, false);
        acc[mi][ni] = __builtin_amdgcn_wmma_f32_16x16x32_bf16(
            false, ah[mi], false, bl[ni], (short)0, acc[mi][ni], false, false);
        acc[mi][ni] = __builtin_amdgcn_wmma_f32_16x16x32_bf16(
            false, al[mi], false, bh[ni], (short)0, acc[mi][ni], false, false);
      }
    }
  }

  // ---- epilogue ----
  const float* bptr = bias + (size_t)v * DIM_TO;
  float bval[4];
  #pragma unroll
  for (int ni = 0; ni < 4; ++ni)
    bval[ni] = (float)TWO_D * bptr[wn + ni * 16 + ln];

  float* og = out + (size_t)v * BATCH * DIM_TO;
  #pragma unroll
  for (int mi = 0; mi < 2; ++mi) {
    #pragma unroll
    for (int ni = 0; ni < 4; ++ni) {
      const int col = wn + ni * 16 + ln;
      #pragma unroll
      for (int r = 0; r < 8; ++r) {
        const int row = m0 + wm + mi * 16 + h * 8 + r;
        og[(size_t)row * DIM_TO + col] = softplus_f(acc[mi][ni][r] + bval[ni]);
      }
    }
  }
}

// ---- fallback (no workspace): fully fused, 128x128 block, 256 threads ----
__global__ __launch_bounds__(256)
void gemm_softplus_fused(const float* __restrict__ x,
                         const float* __restrict__ w,
                         const float* __restrict__ bias,
                         float* __restrict__ out) {
  constexpr int BM = 128, BN = 128;
  __shared__ __align__(16) bf16_t sAhi[BM * LDSB];
  __shared__ __align__(16) bf16_t sAlo[BM * LDSB];
  __shared__ __align__(16) bf16_t sBhi[BN * LDSB];
  __shared__ __align__(16) bf16_t sBlo[BN * LDSB];

  const int t  = threadIdx.x;
  const int v  = blockIdx.z;
  const int m0 = blockIdx.x * BM;
  const int o0 = blockIdx.y * BN;

  const float* xg = x + ((size_t)v * BATCH + m0) * TWO_D;
  const float* wg = w + (size_t)v * TWO_D * DIM_TO + o0;

  const int lane = t & 31;
  const int h  = lane >> 4, ln = lane & 15;
  const int wv = t >> 5;
  const int wm = (wv >> 1) * 32;
  const int wn = (wv & 1) * 64;
  const int ar = t >> 3;
  const int ac = (t & 7) * 4;
  const int ob = (t & 7) * 16;

  v8f acc[2][4] = {};

  for (int k0 = 0; k0 < TWO_D; k0 += BK) {
    __syncthreads();
    #pragma unroll
    for (int p = 0; p < 4; ++p) {
      const int row = p * 32 + ar;
      const float4 f = *(const float4*)(xg + (size_t)row * TWO_D + (k0 + ac));
      const float fx[4] = {f.x, f.y, f.z, f.w};
      v4bf hi4, lo4;
      #pragma unroll
      for (int j = 0; j < 4; ++j) {
        const bf16_t hh = f2bf(fx[j]);
        hi4[j] = hh;
        lo4[j] = f2bf(fx[j] - bf2f(hh));
      }
      *(v4bf*)&sAhi[row * LDSB + ac] = hi4;
      *(v4bf*)&sAlo[row * LDSB + ac] = lo4;
    }
    {
      const float* wrow = wg + (size_t)(k0 + ar) * DIM_TO + ob;
      #pragma unroll
      for (int j4 = 0; j4 < 16; j4 += 4) {
        const float4 f = *(const float4*)(wrow + j4);
        const float fx[4] = {f.x, f.y, f.z, f.w};
        #pragma unroll
        for (int j = 0; j < 4; ++j) {
          const bf16_t hh = f2bf(fx[j]);
          sBhi[(ob + j4 + j) * LDSB + ar] = hh;
          sBlo[(ob + j4 + j) * LDSB + ar] = f2bf(fx[j] - bf2f(hh));
        }
      }
    }
    __syncthreads();

    v16bf ah[2], al[2], bh[4], bl[4];
    #pragma unroll
    for (int mi = 0; mi < 2; ++mi) {
      const int row = wm + mi * 16 + ln;
      const bf16_t* ph = &sAhi[row * LDSB];
      const bf16_t* pl = &sAlo[row * LDSB];
      ah[mi] = cat8(*(const v8bf*)(ph + h * 8), *(const v8bf*)(ph + 16 + h * 8));
      al[mi] = cat8(*(const v8bf*)(pl + h * 8), *(const v8bf*)(pl + 16 + h * 8));
    }
    #pragma unroll
    for (int ni = 0; ni < 4; ++ni) {
      const int col = wn + ni * 16 + ln;
      const bf16_t* ph = &sBhi[col * LDSB];
      const bf16_t* pl = &sBlo[col * LDSB];
      bh[ni] = cat8(*(const v8bf*)(ph + h * 8), *(const v8bf*)(ph + 16 + h * 8));
      bl[ni] = cat8(*(const v8bf*)(pl + h * 8), *(const v8bf*)(pl + 16 + h * 8));
    }
    #pragma unroll
    for (int mi = 0; mi < 2; ++mi) {
      #pragma unroll
      for (int ni = 0; ni < 4; ++ni) {
        acc[mi][ni] = __builtin_amdgcn_wmma_f32_16x16x32_bf16(
            false, ah[mi], false, bh[ni], (short)0, acc[mi][ni], false, false);
        acc[mi][ni] = __builtin_amdgcn_wmma_f32_16x16x32_bf16(
            false, ah[mi], false, bl[ni], (short)0, acc[mi][ni], false, false);
        acc[mi][ni] = __builtin_amdgcn_wmma_f32_16x16x32_bf16(
            false, al[mi], false, bh[ni], (short)0, acc[mi][ni], false, false);
      }
    }
  }

  const float* bptr = bias + (size_t)v * DIM_TO + o0;
  float bval[4];
  #pragma unroll
  for (int ni = 0; ni < 4; ++ni)
    bval[ni] = (float)TWO_D * bptr[wn + ni * 16 + ln];

  float* og = out + (size_t)v * BATCH * DIM_TO;
  #pragma unroll
  for (int mi = 0; mi < 2; ++mi) {
    #pragma unroll
    for (int ni = 0; ni < 4; ++ni) {
      const int col = o0 + wn + ni * 16 + ln;
      #pragma unroll
      for (int r = 0; r < 8; ++r) {
        const int row = m0 + wm + mi * 16 + h * 8 + r;
        og[(size_t)row * DIM_TO + col] = softplus_f(acc[mi][ni][r] + bval[ni]);
      }
    }
  }
}

extern "C" void kernel_launch(void* const* d_in, const int* in_sizes, int n_in,
                              void* d_out, int out_size, void* d_ws, size_t ws_size,
                              hipStream_t stream) {
  const float* x    = (const float*)d_in[0];
  const float* w    = (const float*)d_in[1];
  const float* bias = (const float*)d_in[2];
  float* out = (float*)d_out;

  const size_t welems = (size_t)NUM_VARI * TWO_D * DIM_TO;
  if (ws_size >= 2 * welems * sizeof(bf16_t)) {
    bf16_t* whi = (bf16_t*)d_ws;
    bf16_t* wlo = whi + welems;
    convert_w_kernel<<<(NUM_VARI * DIM_TO) / 256, 256, 0, stream>>>(w, whi, wlo);
    dim3 grid(BATCH / 128, 1, NUM_VARI);                 // (32, 1, 32)
    gemm_softplus_fast<<<grid, 512, 0, stream>>>(x, whi, wlo, bias, out);
  } else {
    dim3 grid(BATCH / 128, DIM_TO / 128, NUM_VARI);      // (32, 2, 32)
    gemm_softplus_fused<<<grid, 256, 0, stream>>>(x, w, bias, out);
  }
}
